// StandardAttention_22376779612873
// MI455X (gfx1250) — matrix operand order repaired and checked by reading, not compiled, not run on hardware
//
#include <hip/hip_runtime.h>
#include <hip/hip_bf16.h>

// ---------------------------------------------------------------------------
// StandardAttention for MI455X (gfx1250, wave32, WMMA + async-to-LDS)
//  X[B,N,4096] @ Wq/Wk/Wv -> RoPE -> causal GQA flash attention -> @ Wo
//  B=2, N=2048, HIDDEN=4096, H=32, KVH=8, D=128
// ---------------------------------------------------------------------------

typedef _Float16 half_t;
typedef _Float16 v16h __attribute__((ext_vector_type(16)));
typedef _Float16 v8h  __attribute__((ext_vector_type(8)));
typedef float    v8f  __attribute__((ext_vector_type(8)));
typedef int      v4i  __attribute__((ext_vector_type(4)));

union Frag16 { v16h v; v8h c[2]; };

#define AS1 __attribute__((address_space(1)))
#define AS3 __attribute__((address_space(3)))

#if defined(__gfx1250__) && __has_builtin(__builtin_amdgcn_global_load_async_to_lds_b128)
#define HAVE_ASYNC 1
#else
#define HAVE_ASYNC 0
#endif

// Device-pass-only compile probe: tells us in stderr whether the async path
// actually lowered (host pass does not define __AMDGCN__).
#if defined(__AMDGCN__)
#if HAVE_ASYNC
#warning "CDNA5 probe: global_load_async_to_lds_b128 AVAILABLE (async path active)"
#else
#warning "CDNA5 probe: async-to-LDS builtins MISSING (sync fallback active)"
#endif
#endif

// 16-byte global -> LDS copy; async (ASYNCcnt-tracked) when available.
__device__ __forceinline__ void async_copy16(const half_t* g, half_t* l) {
#if HAVE_ASYNC
  __builtin_amdgcn_global_load_async_to_lds_b128(
      (AS1 v4i*)g, (AS3 v4i*)l, 0, 0);
#else
  *(v8h*)l = *(const v8h*)g;
#endif
}

template <int Nw>
__device__ __forceinline__ void wait_async() {
#if HAVE_ASYNC
#if __has_builtin(__builtin_amdgcn_s_wait_asynccnt)
  __builtin_amdgcn_s_wait_asynccnt(Nw);
#else
  asm volatile("s_wait_asynccnt %0" ::"i"(Nw) : "memory");
#endif
#endif
}

// A-matrix 16x32 f16 fragment: lanes 0-15 hold K=0..15,
// lanes 16-31 hold K=16..31 (contiguous per lane, matching B).
__device__ __forceinline__ v16h load_afrag(const half_t* row, int hi) {
  Frag16 f;
  f.c[0] = *(const v8h*)(row + (hi ? 16 : 0));
  f.c[1] = *(const v8h*)(row + (hi ? 24 : 8));
  return f.v;
}

// B-matrix 32x16 f16 fragment: lane holds one column; lanes 0-15 hold K=0..15,
// lanes 16-31 hold K=16..31 (contiguous per lane).
__device__ __forceinline__ v16h load_bfrag(const half_t* col, int hi) {
  Frag16 f;
  const half_t* p = col + (hi ? 16 : 0);
  f.c[0] = *(const v8h*)(p);
  f.c[1] = *(const v8h*)(p + 8);
  return f.v;
}

__device__ __forceinline__ v8f wmma_f16(v16h a, v16h b, v8f c) {
  return __builtin_amdgcn_wmma_f32_16x16x32_f16(false, a, false, b,
                                                (short)0, c, false, false);
}

// ---------------------------------------------------------------------------
// f32 -> f16 conversion (row-major, for activations)
// ---------------------------------------------------------------------------
__global__ __launch_bounds__(256) void f32_to_f16_kernel(
    const float* __restrict__ src, half_t* __restrict__ dst, size_t n) {
  size_t i = (size_t)blockIdx.x * 256 + threadIdx.x;
  if (i < n) dst[i] = (half_t)src[i];
}

// ---------------------------------------------------------------------------
// Tiled transpose + convert: src f32 [K,N] -> dst f16 [N,K] (both coalesced)
// ---------------------------------------------------------------------------
__global__ __launch_bounds__(256) void transpose_cvt_kernel(
    const float* __restrict__ src, half_t* __restrict__ dst, int K, int N) {
  __shared__ half_t tile[32][33];
  int k0 = blockIdx.y * 32;
  int n0 = blockIdx.x * 32;
  int tx = threadIdx.x & 31;
  int ty = threadIdx.x >> 5;   // 0..7
#pragma unroll
  for (int i = 0; i < 32; i += 8)
    tile[ty + i][tx] = (half_t)src[(size_t)(k0 + ty + i) * N + n0 + tx];
  __syncthreads();
#pragma unroll
  for (int i = 0; i < 32; i += 8)
    dst[(size_t)(n0 + ty + i) * K + k0 + tx] = tile[tx][ty + i];
}

// ---------------------------------------------------------------------------
// f16 WMMA GEMM with async-to-LDS double buffering:
//   C[M,N] = A[M,K] * Bt[N,K]^T   (Bt is the pre-transposed weight)
// Block tile 128x64, 128 threads (4 waves); wave tile 32x64 (8 WMMA / K-step).
// ---------------------------------------------------------------------------
#define LPAD 8

template <bool OUT_F32>
__global__ __launch_bounds__(128) void gemm16_kernel(
    const half_t* __restrict__ A, const half_t* __restrict__ Bt,
    void* __restrict__ Cout, int M, int N, int K) {
  __shared__ half_t As[2][128][32 + LPAD];
  __shared__ half_t Bs[2][64][32 + LPAD];

  const int tid  = threadIdx.x;
  const int lane = tid & 31;
  const int wave = tid >> 5;
  const int r    = lane & 15;
  const int hi   = lane >> 4;
  const int mb   = blockIdx.y * 128;
  const int nb   = blockIdx.x * 64;

  // 6 async 16B copies per thread per K-tile (4 for A row, 2 for Bt row-half)
  auto stage = [&](int buf, int k0) {
    const half_t* asrc = A + (size_t)(mb + tid) * K + k0;
#pragma unroll
    for (int j = 0; j < 4; ++j)
      async_copy16(asrc + 8 * j, &As[buf][tid][8 * j]);
    int brow = tid >> 1;
    int bcol = (tid & 1) * 16;
    const half_t* bsrc = Bt + (size_t)(nb + brow) * K + k0 + bcol;
#pragma unroll
    for (int j = 0; j < 2; ++j)
      async_copy16(bsrc + 8 * j, &Bs[buf][brow][bcol + 8 * j]);
  };

  v8f acc0[4] = {}, acc1[4] = {};

  const int nsteps = K / 32;
  stage(0, 0);
  for (int s = 0; s < nsteps; ++s) {
    const int cur = s & 1;
    const bool has_next = (s + 1) < nsteps;
    if (has_next) stage(cur ^ 1, (s + 1) * 32);
    // async loads complete in order: waiting to <=6 outstanding guarantees the
    // previous tile's 6 copies have landed in LDS.
    if (has_next) wait_async<6>(); else wait_async<0>();
    __syncthreads();

    v16h a0 = load_afrag(&As[cur][wave * 32 + r][0], hi);
    v16h a1 = load_afrag(&As[cur][wave * 32 + 16 + r][0], hi);
#pragma unroll
    for (int nt = 0; nt < 4; ++nt) {
      v16h bf = load_bfrag(&Bs[cur][nt * 16 + r][0], hi);
      acc0[nt] = wmma_f16(a0, bf, acc0[nt]);
      acc1[nt] = wmma_f16(a1, bf, acc1[nt]);
    }
    __syncthreads();
  }

  // C/D layout: lane half hi, VGPR p -> (row p + 8*hi, col r)
#pragma unroll
  for (int g = 0; g < 2; ++g) {
    const v8f* acc = g ? acc1 : acc0;
#pragma unroll
    for (int nt = 0; nt < 4; ++nt) {
      int col = nb + nt * 16 + r;
#pragma unroll
      for (int p = 0; p < 8; ++p) {
        size_t row = (size_t)(mb + wave * 32 + g * 16 + p + 8 * hi);
        if (OUT_F32)
          ((float*)Cout)[row * N + col] = acc[nt][p];
        else
          ((half_t*)Cout)[row * N + col] = (half_t)acc[nt][p];
      }
    }
  }
}

// ---------------------------------------------------------------------------
// RoPE: X[tok, head, 128]; pair rotation (d, d+64), angle = pos*theta^(-2d/128)
// ---------------------------------------------------------------------------
__global__ __launch_bounds__(256) void rope_kernel(
    half_t* __restrict__ X, const int* __restrict__ pos, int heads, int ntok) {
  int idx = blockIdx.x * 256 + threadIdx.x;
  int total = ntok * heads * 64;
  if (idx >= total) return;
  int d   = idx & 63;
  int hh  = (idx >> 6) % heads;
  int tok = idx / (64 * heads);
  float p = (float)pos[tok];
  float a = p * __expf(-(float)d * 0.14391156f);  // 10000^(-2d/128)
  float c = __cosf(a), s = __sinf(a);
  half_t* base = X + ((size_t)tok * heads + hh) * 128;
  float x0 = (float)base[d];
  float x1 = (float)base[d + 64];
  base[d]      = (half_t)(x0 * c - x1 * s);
  base[d + 64] = (half_t)(x1 * c + x0 * s);
}

// ---------------------------------------------------------------------------
// Causal GQA flash attention: one wave per 16-query tile per (b, h).
// Q/K/V stored [B,N,heads,128] f16. Online softmax over 32-key blocks.
// ---------------------------------------------------------------------------
__global__ __launch_bounds__(32) void attn_kernel(
    const half_t* __restrict__ Q, const half_t* __restrict__ Kx,
    const half_t* __restrict__ V, half_t* __restrict__ O) {
  constexpr int NTOK = 2048, H = 32, KVH = 8, D = 128;
  __shared__ half_t Pt[16][32 + LPAD];    // P tile, row-major [q][key]
  __shared__ half_t Vt[128][32 + LPAD];   // V tile transposed [d][key]

  const int lane = threadIdx.x;
  const int r  = lane & 15;
  const int hi = lane >> 4;
  int tile  = blockIdx.x;
  int qt    = tile & 127;          // N/16 tiles
  int h     = (tile >> 7) & 31;
  int b     = tile >> 12;
  int kv    = h >> 2;              // rep = H/KVH = 4
  int qbase = qt * 16;

  // Q fragments for this wave's 16 rows, covering D=128 (4 x K32 slices)
  const half_t* qrow = Q + (((size_t)b * NTOK + qbase + r) * H + h) * D;
  v16h qf[4];
#pragma unroll
  for (int t = 0; t < 4; ++t) qf[t] = load_afrag(qrow + t * 32, hi);

  v8f o[8] = {};                   // 16x128 output accumulator
  float mrow[8], lrow[8];
#pragma unroll
  for (int p = 0; p < 8; ++p) { mrow[p] = -1e30f; lrow[p] = 0.0f; }

  const float scale = 0.08838834764831845f;   // 1/sqrt(128)

  int nkb = (qbase + 15) / 32 + 1;
  for (int kb = 0; kb < nkb; ++kb) {
    int key0 = kb * 32;

    // Stage V block (32 keys x 128) transposed into LDS
    for (int i = lane; i < 32 * 16; i += 32) {
      int kk = i >> 4;             // key within block
      int d8 = (i & 15) * 8;       // d chunk
      const half_t* vsrc =
          V + (((size_t)b * NTOK + key0 + kk) * KVH + kv) * D + d8;
      v8h x = *(const v8h*)vsrc;
#pragma unroll
      for (int j = 0; j < 8; ++j) Vt[d8 + j][kk] = x[j];
    }

    // Preload all 8 K fragments, then run the WMMA chain
    v16h kf[8];
#pragma unroll
    for (int t = 0; t < 4; ++t) {
      const half_t* kr0 =
          Kx + (((size_t)b * NTOK + key0 + r) * KVH + kv) * D + t * 32;
      const half_t* kr1 =
          Kx + (((size_t)b * NTOK + key0 + 16 + r) * KVH + kv) * D + t * 32;
      kf[2 * t]     = load_bfrag(kr0, hi);
      kf[2 * t + 1] = load_bfrag(kr1, hi);
    }
    v8f s0 = {}, s1 = {};
#pragma unroll
    for (int t = 0; t < 4; ++t) {
      s0 = wmma_f16(qf[t], kf[2 * t], s0);
      s1 = wmma_f16(qf[t], kf[2 * t + 1], s1);
    }

    // Causal mask + online softmax (rows striped across 16-lane halves)
    float mnew[8];
#pragma unroll
    for (int p = 0; p < 8; ++p) {
      int q = qbase + p + 8 * hi;
      float x0 = s0[p] * scale;
      float x1 = s1[p] * scale;
      if (key0 + r > q)      x0 = -1e30f;
      if (key0 + 16 + r > q) x1 = -1e30f;
      s0[p] = x0; s1[p] = x1;
      float mx = fmaxf(x0, x1);
#pragma unroll
      for (int msk = 8; msk >= 1; msk >>= 1)
        mx = fmaxf(mx, __shfl_xor(mx, msk, 32));
      mnew[p] = fmaxf(mrow[p], mx);
    }

#pragma unroll
    for (int p = 0; p < 8; ++p) {
      float e0 = __expf(s0[p] - mnew[p]);
      float e1 = __expf(s1[p] - mnew[p]);
      float rs = e0 + e1;
#pragma unroll
      for (int msk = 8; msk >= 1; msk >>= 1) rs += __shfl_xor(rs, msk, 32);
      float corr = __expf(mrow[p] - mnew[p]);
      lrow[p] = lrow[p] * corr + rs;
      mrow[p] = mnew[p];
#pragma unroll
      for (int dt = 0; dt < 8; ++dt) o[dt][p] *= corr;
      Pt[p + 8 * hi][r]      = (half_t)e0;
      Pt[p + 8 * hi][16 + r] = (half_t)e1;
    }
    __syncthreads();

    // O += P * V  (P re-laid C->A layout via LDS roundtrip)
    v16h pf = load_afrag(&Pt[r][0], hi);
    v16h vf[8];
#pragma unroll
    for (int dt = 0; dt < 8; ++dt)
      vf[dt] = load_bfrag(&Vt[dt * 16 + r][0], hi);
#pragma unroll
    for (int dt = 0; dt < 8; ++dt) o[dt] = wmma_f16(pf, vf[dt], o[dt]);
    __syncthreads();
  }

  // Normalize and write [B,N,H,D] f16
#pragma unroll
  for (int p = 0; p < 8; ++p) {
    float inv = 1.0f / lrow[p];
    int q = qbase + p + 8 * hi;
    half_t* dst = O + (((size_t)b * NTOK + q) * H + h) * D;
#pragma unroll
    for (int dt = 0; dt < 8; ++dt)
      dst[dt * 16 + r] = (half_t)(o[dt][p] * inv);
  }
}

// ---------------------------------------------------------------------------
// Host-side orchestration
// ---------------------------------------------------------------------------
extern "C" void kernel_launch(void* const* d_in, const int* in_sizes, int n_in,
                              void* d_out, int out_size, void* d_ws,
                              size_t ws_size, hipStream_t stream) {
  (void)in_sizes; (void)n_in; (void)out_size; (void)ws_size;

  const float* X  = (const float*)d_in[0];
  const int*  pos = (const int*)d_in[1];
  const float* Wq = (const float*)d_in[2];
  const float* Wk = (const float*)d_in[3];
  const float* Wv = (const float*)d_in[4];
  const float* Wo = (const float*)d_in[5];
  float* out = (float*)d_out;

  const int B = 2, N = 2048, HID = 4096, H = 32, KVH = 8;
  const size_t M  = (size_t)B * N;   // 4096 token rows
  const size_t QO = 4096;            // H*D
  const size_t KO = 1024;            // KVH*D

  half_t* p   = (half_t*)d_ws;
  half_t* Xh  = p; p += M * HID;
  half_t* WqT = p; p += QO * HID;    // [QO][HID]
  half_t* WkT = p; p += KO * HID;    // [KO][HID]
  half_t* WvT = p; p += KO * HID;
  half_t* WoT = p; p += (size_t)HID * QO;  // [HID][QO]
  half_t* Qh  = p; p += M * QO;
  half_t* Kh  = p; p += M * KO;
  half_t* Vh  = p; p += M * KO;
  half_t* Ah  = p; p += M * QO;

  f32_to_f16_kernel<<<dim3((unsigned)((M * HID + 255) / 256)), dim3(256), 0,
                      stream>>>(X, Xh, M * HID);
  // Weight transposes: src [K,N] f32 -> dst [N,K] f16
  transpose_cvt_kernel<<<dim3(QO / 32, HID / 32), dim3(256), 0, stream>>>(
      Wq, WqT, HID, (int)QO);
  transpose_cvt_kernel<<<dim3(KO / 32, HID / 32), dim3(256), 0, stream>>>(
      Wk, WkT, HID, (int)KO);
  transpose_cvt_kernel<<<dim3(KO / 32, HID / 32), dim3(256), 0, stream>>>(
      Wv, WvT, HID, (int)KO);
  transpose_cvt_kernel<<<dim3(HID / 32, QO / 32), dim3(256), 0, stream>>>(
      Wo, WoT, (int)QO, HID);

  dim3 gblk(128);
  // Q/K/V projections (f16 out): grid (N/64, M/128)
  gemm16_kernel<false><<<dim3(QO / 64, M / 128), gblk, 0, stream>>>(
      Xh, WqT, Qh, (int)M, (int)QO, HID);
  gemm16_kernel<false><<<dim3(KO / 64, M / 128), gblk, 0, stream>>>(
      Xh, WkT, Kh, (int)M, (int)KO, HID);
  gemm16_kernel<false><<<dim3(KO / 64, M / 128), gblk, 0, stream>>>(
      Xh, WvT, Vh, (int)M, (int)KO, HID);

  // RoPE on Q and K
  rope_kernel<<<dim3((unsigned)((M * H * 64 + 255) / 256)), dim3(256), 0,
                stream>>>(Qh, pos, H, (int)M);
  rope_kernel<<<dim3((unsigned)((M * KVH * 64 + 255) / 256)), dim3(256), 0,
                stream>>>(Kh, pos, KVH, (int)M);

  // Flash attention: B * H * (N/16) waves
  attn_kernel<<<dim3(B * H * (N / 16)), dim3(32), 0, stream>>>(Qh, Kh, Vh, Ah);

  // Output projection (f32 out)
  gemm16_kernel<true><<<dim3(HID / 64, M / 128), gblk, 0, stream>>>(
      Ah, WoT, out, (int)M, HID, (int)QO);
}
